// DynamicContactNet_28269474742396
// MI455X (gfx1250) — compile-verified, matrix operand order checked by reading, not run on hardware
//
#include <hip/hip_runtime.h>
#include <hip/hip_bf16.h>
#include <math.h>

typedef __attribute__((ext_vector_type(16))) _Float16 v16h;
typedef __attribute__((ext_vector_type(8)))  _Float16 v8h;
typedef __attribute__((ext_vector_type(8)))  float    v8f;

#define DEV __device__ __forceinline__

DEV int lane_id() { return threadIdx.x & 31; }
DEV int d_row(int r) { return r + ((lane_id() >> 4) << 3); }  // C/D frag: M per VGPR r
DEV int d_col()      { return lane_id() & 15; }               // C/D frag: N per lane

// branchless native transcendentals (v_exp_f32 / v_rcp_f32)
DEV float fast_rcp(float x)  { return __builtin_amdgcn_rcpf(x); }
DEV float fast_tanh(float x) { return 1.f - 2.f * fast_rcp(1.f + __expf(2.f * x)); }
DEV float fast_sigmoid(float x) { return fast_rcp(1.f + __expf(-x)); }

// Fragment gather from K-contiguous storage (row-major [M][ldk] for A, or
// [N][ldk] for B since the B lane layout is the transposed read of A's).
// Per ISA 7.12.2 the 16 f16 of a lane are two contiguous 8-element runs
// {kb..kb+7} and {kb+16..kb+23}, kb = (lane>=16 ? 8 : 0)  ->  2x b128 loads.
DEV v16h frag16(const _Float16* base, int ld, int r0, int k0) {
  int lane = lane_id();
  const _Float16* p = base + (size_t)(r0 + (lane & 15)) * ld + k0 + ((lane >> 4) << 3);
  v8h lo = *(const v8h*)p;
  v8h hi = *(const v8h*)(p + 16);
  return __builtin_shufflevector(lo, hi, 0, 1, 2, 3, 4, 5, 6, 7,
                                 8, 9, 10, 11, 12, 13, 14, 15);
}

DEV v8f wmma_f16(v16h a, v16h b, v8f c) {
  return __builtin_amdgcn_wmma_f32_16x16x32_f16(false, a, false, b, (short)0, c,
                                                false, false);
}

// ---------------------------------------------------------------- prep kernels

// fp32 [K][N] -> f16 transposed [N][K] (K-contiguous for b128 fragment loads)
__global__ void cvt_transpose_f16(const float* __restrict__ s, _Float16* __restrict__ d,
                                  int K, int N) {
  int t = blockIdx.x * blockDim.x + threadIdx.x;
  if (t < K * N) {
    int k = t / N, n = t - k * N;
    d[(size_t)n * K + k] = (_Float16)s[t];
  }
}

// pi = single @ film_W1[:256], pj = single @ film_W1[256:]   (512 x 128 each)
__global__ void pipj_kernel(const float* __restrict__ single, const float* __restrict__ W1,
                            float* __restrict__ pi, float* __restrict__ pj) {
  int t = blockIdx.x * blockDim.x + threadIdx.x;   // 131072 threads
  int hsel = t >> 16;
  int idx  = t & 65535;
  int tok  = idx >> 7;
  int dch  = idx & 127;
  const float* x = single + (size_t)tok * 256;
  const float* w = W1 + (size_t)hsel * 256 * 128 + dch;
  float s = 0.f;
#pragma unroll 8
  for (int k = 0; k < 256; ++k) s += x[k] * w[(size_t)k * 128];
  (hsel ? pj : pi)[(size_t)tok * 128 + dch] = s;
}

// --------------------------------------------- FiLM + reduce MLP (fused, WMMA)
// grid 4096 = 512 i * 8 j-tiles; block 128 (4 waves, 16 tokens each)
// weights in LDS are [N][K] f16 (K-contiguous).

__global__ __launch_bounds__(128) void film_reduce_kernel(
    const float* __restrict__ pi, const float* __restrict__ pj,
    const float* __restrict__ fb1, const _Float16* __restrict__ W2t,
    const float* __restrict__ fb2, const float* __restrict__ pair,
    const _Float16* __restrict__ rW1t, const float* __restrict__ rb1,
    const _Float16* __restrict__ rW2t, const float* __restrict__ rb2,
    _Float16* __restrict__ p2out) {
  extern __shared__ char smem[];
  _Float16* sW2  = (_Float16*)smem;        // [256][128]
  _Float16* sRW1 = sW2 + 256 * 128;        // [64][128]
  _Float16* sRW2 = sRW1 + 64 * 128;        // [32][64]
  _Float16* sH   = sRW2 + 32 * 64;         // [64][136]  h tile (later: z staging)
  _Float16* sP2  = sH + 64 * 136;          // [64][136]  pair2 tile
  _Float16* sG   = sP2 + 64 * 136;         // [64][72]   gelu tile
  float*    sB2  = (float*)(sG + 64 * 72); // 256 (film_b2)
  float*    sRB1 = sB2 + 256;              // 64
  float*    sRB2 = sRB1 + 64;              // 32

  const int tid = threadIdx.x;
  const int wave = tid >> 5;
  const int i  = blockIdx.x >> 3;
  const int j0 = (blockIdx.x & 7) << 6;

  // cooperative weight loads (b128)
  {
    const uint4* s0 = (const uint4*)W2t;
    uint4* d0 = (uint4*)sW2;
    for (int idx = tid; idx < 256 * 128 / 8; idx += 128) d0[idx] = s0[idx];
    const uint4* s1 = (const uint4*)rW1t;
    uint4* d1 = (uint4*)sRW1;
    for (int idx = tid; idx < 64 * 128 / 8; idx += 128) d1[idx] = s1[idx];
    const uint4* s2 = (const uint4*)rW2t;
    uint4* d2 = (uint4*)sRW2;
    for (int idx = tid; idx < 32 * 64 / 8; idx += 128) d2[idx] = s2[idx];
    for (int idx = tid; idx < 256; idx += 128) sB2[idx] = fb2[idx];
    if (tid < 64) sRB1[tid] = rb1[tid];
    if (tid < 32) sRB2[tid] = rb2[tid];
  }
  // h = relu(pi[i] + pj[j] + b1), cooperative + coalesced
  {
    const float* pii = pi + (size_t)i * 128;
    for (int idx = tid; idx < 64 * 128; idx += 128) {
      int m = idx >> 7, k = idx & 127;
      float v = pii[k] + pj[(size_t)(j0 + m) * 128 + k] + fb1[k];
      sH[m * 136 + k] = (_Float16)fmaxf(v, 0.f);
    }
  }
  __syncthreads();

  const int nc = d_col();
  // hoist A fragments (K=128 -> 4 chunks)
  v16h afr[4];
#pragma unroll
  for (int c4 = 0; c4 < 4; ++c4) afr[c4] = frag16(sH, 136, wave * 16, c4 * 32);

  // GEMM1: x = h @ film_W2 + b2, fused FiLM
#pragma unroll 1
  for (int nt = 0; nt < 8; ++nt) {
    v8f accG, accB;
    const int cg = nt * 16 + nc;              // channel 0..127
#pragma unroll
    for (int r = 0; r < 8; ++r) { accG[r] = sB2[cg]; accB[r] = sB2[cg + 128]; }
#pragma unroll
    for (int c4 = 0; c4 < 4; ++c4) {
      accG = wmma_f16(afr[c4], frag16(sW2, 128, nt * 16, c4 * 32), accG);
      accB = wmma_f16(afr[c4], frag16(sW2, 128, 128 + nt * 16, c4 * 32), accB);
    }
#pragma unroll
    for (int r = 0; r < 8; ++r) {
      int m = d_row(r);
      float gam = 1.f + 0.2f * fast_tanh(accG[r]);
      float bet = 0.2f * fast_tanh(accB[r]);
      float pv = pair[(((size_t)i * 512) + j0 + wave * 16 + m) * 128 + cg];
      sP2[(wave * 16 + m) * 136 + cg] = (_Float16)(gam * pv + bet);
    }
  }
  __syncthreads();

  // GEMM2: g = gelu(pair2 @ red_W1 + rb1)   (K=128, N=64)
#pragma unroll 1
  for (int nt = 0; nt < 4; ++nt) {
    v8f acc;
    int n = nt * 16 + nc;
#pragma unroll
    for (int r = 0; r < 8; ++r) acc[r] = sRB1[n];
#pragma unroll
    for (int c4 = 0; c4 < 4; ++c4)
      acc = wmma_f16(frag16(sP2, 136, wave * 16, c4 * 32),
                     frag16(sRW1, 128, nt * 16, c4 * 32), acc);
#pragma unroll
    for (int r = 0; r < 8; ++r) {
      float x = acc[r];
      float ge = 0.5f * x * (1.f + erff(x * 0.70710678118654752f));
      sG[(wave * 16 + d_row(r)) * 72 + n] = (_Float16)ge;
    }
  }
  __syncthreads();

  // GEMM3: z = g @ red_W2 + rb2  (K=64, N=32), stage into sH (free now)
#pragma unroll 1
  for (int nt = 0; nt < 2; ++nt) {
    v8f acc;
    int n = nt * 16 + nc;
#pragma unroll
    for (int r = 0; r < 8; ++r) acc[r] = sRB2[n];
#pragma unroll
    for (int c4 = 0; c4 < 2; ++c4)
      acc = wmma_f16(frag16(sG, 72, wave * 16, c4 * 32),
                     frag16(sRW2, 64, nt * 16, c4 * 32), acc);
#pragma unroll
    for (int r = 0; r < 8; ++r)
      sH[(wave * 16 + d_row(r)) * 136 + n] = (_Float16)acc[r];
  }
  __syncthreads();

  // vectorized store: 64 rows x 32 f16 (4x b128 per row)
  for (int idx = tid; idx < 64 * 4; idx += 128) {
    int m = idx >> 2, c = idx & 3;
    uint4 v = *(const uint4*)(sH + m * 136 + c * 8);
    *(uint4*)(p2out + (((size_t)i * 512) + j0 + m) * 32 + c * 8) = v;
  }
}

// --------------------------------------------- windowed 2-head MHSA (row/col)
// grid 4096 = 512 sequences * 8 windows; block 128

__global__ __launch_bounds__(128) void attn_kernel(
    const _Float16* __restrict__ xin, _Float16* __restrict__ xout,
    const _Float16* __restrict__ qkvWt, const float* __restrict__ qkvb,
    const _Float16* __restrict__ outWt, const float* __restrict__ outb,
    int colpass) {
  extern __shared__ char smem[];
  _Float16* sX  = (_Float16*)smem;         // [64][32] in tile (later: out staging)
  _Float16* sQ  = sX + 64 * 32;            // [2][64][32] (dims 16..31 zero)
  _Float16* sK  = sQ + 2 * 64 * 32;        // [2][64][32] (dims 16..31 zero)
  _Float16* sVt = sK + 2 * 64 * 32;        // [2][16][64] v transposed
  _Float16* sO  = sVt + 2 * 16 * 64;       // [64][32]
  _Float16* sQW = sO + 64 * 32;            // [96][32]
  _Float16* sOW = sQW + 96 * 32;           // [32][32]
  float*    sS  = (float*)(sOW + 32 * 32); // [2][64][64] scores f32
  _Float16* sA  = (_Float16*)(sS + 2 * 64 * 64); // [2][64][64] att f16
  float*    sQB = (float*)(sA + 2 * 64 * 64);    // 96
  float*    sOB = sQB + 96;                       // 32

  const int tid  = threadIdx.x;
  const int wave = tid >> 5;
  const int p = blockIdx.x >> 3;   // sequence index
  const int q = blockIdx.x & 7;    // window index

  // load 64x32 tile, b128 per 8 channels
  for (int idx = tid; idx < 64 * 4; idx += 128) {
    int m = idx >> 2, c = idx & 3;
    size_t gi = colpass ? (size_t)(q * 64 + m) : (size_t)p;
    size_t gj = colpass ? (size_t)p : (size_t)(q * 64 + m);
    ((uint4*)sX)[idx] = *(const uint4*)(xin + (gi * 512 + gj) * 32 + c * 8);
  }
  {
    const uint4* s0 = (const uint4*)qkvWt;
    uint4* d0 = (uint4*)sQW;
    for (int idx = tid; idx < 96 * 32 / 8; idx += 128) d0[idx] = s0[idx];
    const uint4* s1 = (const uint4*)outWt;
    uint4* d1 = (uint4*)sOW;
    for (int idx = tid; idx < 32 * 32 / 8; idx += 128) d1[idx] = s1[idx];
    if (tid < 96) sQB[tid] = qkvb[tid];
    if (tid < 32) sOB[tid] = outb[tid];
  }
  for (int idx = tid; idx < 2 * 64 * 32 / 8; idx += 128) {
    ((uint4*)sQ)[idx] = make_uint4(0, 0, 0, 0);
    ((uint4*)sK)[idx] = make_uint4(0, 0, 0, 0);
  }
  __syncthreads();

  const int nc = d_col();
  // QKV projection (M=16/wave, K=32, N=96), scatter into padded q/k + v^T
  {
    v16h ax = frag16(sX, 32, wave * 16, 0);
#pragma unroll 1
    for (int nt = 0; nt < 6; ++nt) {
      v8f acc;
      int n = nt * 16 + nc;
#pragma unroll
      for (int r = 0; r < 8; ++r) acc[r] = sQB[n];
      acc = wmma_f16(ax, frag16(sQW, 32, nt * 16, 0), acc);
#pragma unroll
      for (int r = 0; r < 8; ++r) {
        int m = wave * 16 + d_row(r);
        float v = acc[r];
        if (n < 32) {
          sQ[((n >> 4) * 64 + m) * 32 + (n & 15)] = (_Float16)(v * 0.25f); // *hd^-0.5
        } else if (n < 64) {
          int nn = n - 32;
          sK[((nn >> 4) * 64 + m) * 32 + (nn & 15)] = (_Float16)v;
        } else {
          int nn = n - 64;
          sVt[((nn >> 4) * 16 + (nn & 15)) * 64 + m] = (_Float16)v;
        }
      }
    }
  }
  __syncthreads();

  // scores = q @ k^T per head (K=32, top 16 dims zero) -> f32 LDS
#pragma unroll 1
  for (int h = 0; h < 2; ++h) {
    v16h aq = frag16(sQ + h * 64 * 32, 32, wave * 16, 0);
#pragma unroll 1
    for (int nt = 0; nt < 4; ++nt) {
      v8f acc;
#pragma unroll
      for (int r = 0; r < 8; ++r) acc[r] = 0.f;
      acc = wmma_f16(aq, frag16(sK + h * 64 * 32, 32, nt * 16, 0), acc);
#pragma unroll
      for (int r = 0; r < 8; ++r)
        sS[(h * 64 + wave * 16 + d_row(r)) * 64 + nt * 16 + nc] = acc[r];
    }
  }
  __syncthreads();

  // softmax: one thread per (head,row); 128 threads == 2*64 rows
  {
    int h = tid >> 6, m = tid & 63;
    float* row = sS + (h * 64 + m) * 64;
    float mx = row[0];
#pragma unroll
    for (int e = 1; e < 64; ++e) mx = fmaxf(mx, row[e]);
    float sum = 0.f;
#pragma unroll
    for (int e = 0; e < 64; ++e) sum += __expf(row[e] - mx);
    float inv = fast_rcp(sum);
    _Float16* arow = sA + (h * 64 + m) * 64;
#pragma unroll
    for (int e = 0; e < 64; ++e) arow[e] = (_Float16)(__expf(row[e] - mx) * inv);
  }
  __syncthreads();

  // o = att @ v per head (K=64, N=16); B from v^T [dim][token]
#pragma unroll 1
  for (int h = 0; h < 2; ++h) {
    v8f acc;
#pragma unroll
    for (int r = 0; r < 8; ++r) acc[r] = 0.f;
#pragma unroll
    for (int c4 = 0; c4 < 2; ++c4)
      acc = wmma_f16(frag16(sA + h * 64 * 64, 64, wave * 16, c4 * 32),
                     frag16(sVt + h * 16 * 64, 64, 0, c4 * 32), acc);
#pragma unroll
    for (int r = 0; r < 8; ++r)
      sO[(wave * 16 + d_row(r)) * 32 + h * 16 + nc] = (_Float16)acc[r];
  }
  __syncthreads();

  // out projection (K=32, N=32), stage into sX, then vectorized store
  {
    v16h ao = frag16(sO, 32, wave * 16, 0);
#pragma unroll 1
    for (int nt = 0; nt < 2; ++nt) {
      v8f acc;
      int n = nt * 16 + nc;
#pragma unroll
      for (int r = 0; r < 8; ++r) acc[r] = sOB[n];
      acc = wmma_f16(ao, frag16(sOW, 32, nt * 16, 0), acc);
#pragma unroll
      for (int r = 0; r < 8; ++r)
        sX[(wave * 16 + d_row(r)) * 32 + n] = (_Float16)acc[r];
    }
  }
  __syncthreads();
  for (int idx = tid; idx < 64 * 4; idx += 128) {
    int m = idx >> 2, c = idx & 3;
    size_t gi = colpass ? (size_t)(q * 64 + m) : (size_t)p;
    size_t gj = colpass ? (size_t)p : (size_t)(q * 64 + m);
    *(uint4*)(xout + (gi * 512 + gj) * 32 + c * 8) = ((const uint4*)sX)[idx];
  }
}

// --------------------------------------------------------- head MLP + sigmoid

__global__ __launch_bounds__(128) void head_kernel(
    const _Float16* __restrict__ p3, const _Float16* __restrict__ hW1t,
    const float* __restrict__ hb1, const float* __restrict__ hW2,
    const float* __restrict__ hb2, float* __restrict__ out) {
  extern __shared__ char smem[];
  _Float16* sW1 = (_Float16*)smem;         // [64][32]
  float*    sT  = (float*)(sW1 + 64 * 32); // [64][68]
  float*    sB1 = sT + 64 * 68;            // 64
  float*    sW2 = sB1 + 64;                // 64

  const int tid = threadIdx.x;
  const int wave = tid >> 5;
  {
    const uint4* s0 = (const uint4*)hW1t;
    uint4* d0 = (uint4*)sW1;
    for (int idx = tid; idx < 64 * 32 / 8; idx += 128) d0[idx] = s0[idx];
    if (tid < 64) sB1[tid] = hb1[tid];
    else if (tid < 128) sW2[tid - 64] = hW2[tid - 64];
  }
  __syncthreads();

  const size_t t0 = (size_t)blockIdx.x * 64;
  v16h ax = frag16(p3 + t0 * 32, 32, wave * 16, 0);   // global b128 pair
  const int nc = d_col();
#pragma unroll 1
  for (int nt = 0; nt < 4; ++nt) {
    v8f acc;
    int n = nt * 16 + nc;
#pragma unroll
    for (int r = 0; r < 8; ++r) acc[r] = sB1[n];
    acc = wmma_f16(ax, frag16(sW1, 32, nt * 16, 0), acc);
#pragma unroll
    for (int r = 0; r < 8; ++r)
      sT[(wave * 16 + d_row(r)) * 68 + n] = fmaxf(acc[r], 0.f);
  }
  __syncthreads();

  if (tid < 64) {
    const float* row = sT + tid * 68;
    float s = hb2[0];
#pragma unroll
    for (int e = 0; e < 64; ++e) s += row[e] * sW2[e];
    out[t0 + tid] = fast_sigmoid(s);
  }
}

// ------------------------------------------------------------------- launcher

extern "C" void kernel_launch(void* const* d_in, const int* in_sizes, int n_in,
                              void* d_out, int out_size, void* d_ws, size_t ws_size,
                              hipStream_t stream) {
  (void)in_sizes; (void)n_in; (void)out_size; (void)ws_size;
  const float* single = (const float*)d_in[0];
  const float* pair   = (const float*)d_in[1];
  const float* filmW1 = (const float*)d_in[2];
  const float* filmb1 = (const float*)d_in[3];
  const float* filmW2 = (const float*)d_in[4];
  const float* filmb2 = (const float*)d_in[5];
  const float* redW1  = (const float*)d_in[6];
  const float* redb1  = (const float*)d_in[7];
  const float* redW2  = (const float*)d_in[8];
  const float* redb2  = (const float*)d_in[9];
  const float* qkvW   = (const float*)d_in[10];
  const float* qkvb   = (const float*)d_in[11];
  const float* outW   = (const float*)d_in[12];
  const float* outb   = (const float*)d_in[13];
  const float* headW1 = (const float*)d_in[14];
  const float* headb1 = (const float*)d_in[15];
  const float* headW2 = (const float*)d_in[16];
  const float* headb2 = (const float*)d_in[17];

  char* ws = (char*)d_ws;
  size_t off = 0;
  auto carve = [&](size_t bytes) -> char* {
    char* r = ws + off;
    off += (bytes + 255) & ~(size_t)255;
    return r;
  };
  float*     pi     = (float*)carve((size_t)512 * 128 * 4);
  float*     pj     = (float*)carve((size_t)512 * 128 * 4);
  _Float16*  W2t    = (_Float16*)carve((size_t)256 * 128 * 2);  // [N=256][K=128]
  _Float16*  rW1t   = (_Float16*)carve((size_t)64 * 128 * 2);   // [64][128]
  _Float16*  rW2t   = (_Float16*)carve((size_t)32 * 64 * 2);    // [32][64]
  _Float16*  qkvWt  = (_Float16*)carve((size_t)96 * 32 * 2);    // [96][32]
  _Float16*  outWt  = (_Float16*)carve((size_t)32 * 32 * 2);    // [32][32]
  _Float16*  hW1t   = (_Float16*)carve((size_t)64 * 32 * 2);    // [64][32]
  _Float16*  p2buf  = (_Float16*)carve((size_t)512 * 512 * 32 * 2);
  _Float16*  rowbuf = (_Float16*)carve((size_t)512 * 512 * 32 * 2);
  _Float16*  p3buf  = (_Float16*)carve((size_t)512 * 512 * 32 * 2);

  cvt_transpose_f16<<<(32768 + 255) / 256, 256, 0, stream>>>(filmW2, W2t, 128, 256);
  cvt_transpose_f16<<<(8192 + 255) / 256, 256, 0, stream>>>(redW1, rW1t, 128, 64);
  cvt_transpose_f16<<<(2048 + 255) / 256, 256, 0, stream>>>(redW2, rW2t, 64, 32);
  cvt_transpose_f16<<<(3072 + 255) / 256, 256, 0, stream>>>(qkvW, qkvWt, 32, 96);
  cvt_transpose_f16<<<(1024 + 255) / 256, 256, 0, stream>>>(outW, outWt, 32, 32);
  cvt_transpose_f16<<<(2048 + 255) / 256, 256, 0, stream>>>(headW1, hW1t, 32, 64);
  pipj_kernel<<<512, 256, 0, stream>>>(single, filmW1, pi, pj);

  size_t shB = (size_t)(256 * 128 + 64 * 128 + 32 * 64 + 64 * 136 + 64 * 136 +
                        64 * 72) * 2 + (size_t)(256 + 64 + 32) * 4;
  film_reduce_kernel<<<4096, 128, shB, stream>>>(pi, pj, filmb1, W2t, filmb2, pair,
                                                 rW1t, redb1, rW2t, redb2, p2buf);

  size_t shA = (size_t)(64 * 32 + 2 * 64 * 32 + 2 * 64 * 32 + 2 * 16 * 64 +
                        64 * 32 + 96 * 32 + 32 * 32) * 2 +
               (size_t)2 * 64 * 64 * 4 + (size_t)2 * 64 * 64 * 2 +
               (size_t)(96 + 32) * 4;
  attn_kernel<<<4096, 128, shA, stream>>>(p2buf, rowbuf, qkvWt, qkvb, outWt, outb, 0);
  attn_kernel<<<4096, 128, shA, stream>>>(rowbuf, p3buf, qkvWt, qkvb, outWt, outb, 1);

  size_t shH = (size_t)64 * 32 * 2 + (size_t)64 * 68 * 4 + (size_t)128 * 4;
  head_kernel<<<4096, 128, shH, stream>>>(p3buf, hW1t, headb1, headW2, headb2,
                                          (float*)d_out);
}
